// MemNet_40931038330965
// MI455X (gfx1250) — compile-verified, matrix operand order checked by reading, not run on hardware
//
#include <hip/hip_runtime.h>
#include <hip/hip_bf16.h>
#include <math.h>

typedef __attribute__((ext_vector_type(16))) _Float16 v16h;
typedef __attribute__((ext_vector_type(8)))  _Float16 v8h;
typedef __attribute__((ext_vector_type(4)))  _Float16 v4h;
typedef __attribute__((ext_vector_type(8)))  float    v8f;
typedef __attribute__((ext_vector_type(4)))  float    v4f;

// ---- model dims (match reference) ----
#define V_    32000
#define E_    512
#define HID_  1024
#define L_    2
#define NH_   8
#define HD_   64
#define B_    4
#define T_    128
#define M_    512
#define S_    1024
#define MH_   4
#define DH_   128
#define TOPK_ 32
#define INTERVAL_ 16
#define DECAY_ 0.99f
#define BT_   128
#define SH_   4
#define SD_   128
#define IFACE_ (4*M_ + 1)   // 2049

#define GF_ACC  1
#define GF_GELU 2
#define GF_BT   4

#define ADDR_READ  0
#define ADDR_WRITE 1

#define KSTEP   64
#define LDS_LD  72          // padded row stride (halves): 144B -> conflict-free b128 reads

__device__ __forceinline__ float gelu_f(float x) {
  const float c = 0.7978845608028654f; // sqrt(2/pi)
  return 0.5f * x * (1.f + tanhf(c * (x + 0.044715f * x * x * x)));
}
__device__ __forceinline__ float sigm_f(float x) { return 1.f / (1.f + __expf(-x)); }

// ---------------------------------------------------------------------------
// f32-in/f32-out GEMM on the f16 WMMA pipe.
//   C[M,N] (+)= act( scale * (A[M,K] @ B[K,N]) + bias )
// Block = 128 threads (4 waves); block tile 64(M) x 64(N); KSTEP=64.
// Wave w owns rows w*16..w*16+15 and ALL 64 columns: 4 accumulators, one A
// fragment reused against 4 B fragments -> 8 WMMAs per k-tile per wave.
// A (64x64) and B (64x64, staged n-major) live in LDS as f16, padded stride.
// REQUIREMENTS (guaranteed by host code): K % 64 == 0, N % 64 == 0, and A has
// at least ceil(M/64)*64 addressable rows. GF_BT: B given as B[n*ldb + k].
// ---------------------------------------------------------------------------
__global__ __launch_bounds__(128)
void k_wmma_gemm(const float* __restrict__ A, const float* __restrict__ B,
                 const float* __restrict__ bias, float* __restrict__ C,
                 int Mdim, int Ndim, int Kdim, int lda, int ldb, int ldc,
                 int flags, float scale)
{
  __shared__ _Float16 Ash[64 * LDS_LD];        // [row][k]
  __shared__ _Float16 Bsh[64 * LDS_LD];        // [n][k]  (n-major!)
  const int tid   = threadIdx.x;
  const int lane  = tid & 31;
  const int wv    = tid >> 5;
  const int row0  = blockIdx.y * 64;
  const int ncol0 = blockIdx.x * 64;
  const int kb    = (lane >> 4) * 8;           // 0 or 8
  const int lrow  = lane & 15;                 // row within wave's 16-row strip
  const int lcol  = lane & 15;                 // col within a 16-col tile

  v8f acc[4] = {v8f{}, v8f{}, v8f{}, v8f{}};

  for (int k0 = 0; k0 < Kdim; k0 += KSTEP) {
    // ---- A tile: 64 rows x 64 k; thread = half a row (8 x v4f -> v4h) ----
    {
      const int r    = tid >> 1;
      const int koff = (tid & 1) * 32;
      const float* ap = A + (size_t)(row0 + r) * lda + k0 + koff;
      #pragma unroll
      for (int u = 0; u < 8; ++u) {
        v4f av = *(const v4f*)(ap + u * 4);
        v4h ah;
        ah[0] = (_Float16)av[0]; ah[1] = (_Float16)av[1];
        ah[2] = (_Float16)av[2]; ah[3] = (_Float16)av[3];
        *(v4h*)(Ash + r * LDS_LD + koff + u * 4) = ah;
      }
      if (k0 + KSTEP < Kdim)
        __builtin_prefetch(ap + KSTEP, 0, 1);
    }
    // ---- B tile: 64 k x 64 n staged as Bsh[n][k] ----
    if (flags & GF_BT) {
      // B[n*ldb + k]: contiguous in k -> straight copy
      const int n    = tid >> 1;
      const int koff = (tid & 1) * 32;
      const float* bp = B + (size_t)(ncol0 + n) * ldb + k0 + koff;
      #pragma unroll
      for (int u = 0; u < 8; ++u) {
        v4f bv = *(const v4f*)(bp + u * 4);
        v4h bh;
        bh[0] = (_Float16)bv[0]; bh[1] = (_Float16)bv[1];
        bh[2] = (_Float16)bv[2]; bh[3] = (_Float16)bv[3];
        *(v4h*)(Bsh + n * LDS_LD + koff + u * 4) = bh;
      }
    } else {
      // B[k*ldb + n]: 4x4 register micro-transpose, 2 blocks per thread
      #pragma unroll
      for (int u = 0; u < 2; ++u) {
        int bi  = tid * 2 + u;                 // 0..255 (16x16 blocks of 4x4)
        int nb  = (bi & 15) * 4;
        int kbq = (bi >> 4) * 4;
        const float* bp = B + (size_t)(k0 + kbq) * ldb + ncol0 + nb;
        v4f r0 = *(const v4f*)(bp);
        v4f r1 = *(const v4f*)(bp + ldb);
        v4f r2 = *(const v4f*)(bp + 2 * (size_t)ldb);
        v4f r3 = *(const v4f*)(bp + 3 * (size_t)ldb);
        #pragma unroll
        for (int i = 0; i < 4; ++i) {
          v4h col;
          col[0] = (_Float16)r0[i]; col[1] = (_Float16)r1[i];
          col[2] = (_Float16)r2[i]; col[3] = (_Float16)r3[i];
          *(v4h*)(Bsh + (size_t)(nb + i) * LDS_LD + kbq) = col;
        }
      }
    }
    __syncthreads();

    // ---- 2 k-halves x 4 n-tiles = 8 WMMAs; A fragment reused 4x ----
    #pragma unroll
    for (int half = 0; half < 2; ++half) {
      const int ko = half * 32;
      const _Float16* apn = Ash + (wv * 16 + lrow) * LDS_LD + ko + kb;
      v8h a0 = *(const v8h*)(apn);
      v8h a1 = *(const v8h*)(apn + 16);
      v16h afrag = __builtin_shufflevector(a0, a1, 0,1,2,3,4,5,6,7,8,9,10,11,12,13,14,15);
      #pragma unroll
      for (int nt = 0; nt < 4; ++nt) {
        const _Float16* bpn = Bsh + (nt * 16 + lcol) * LDS_LD + ko + kb;
        v8h b0 = *(const v8h*)(bpn);
        v8h b1 = *(const v8h*)(bpn + 16);
        v16h bfrag = __builtin_shufflevector(b0, b1, 0,1,2,3,4,5,6,7,8,9,10,11,12,13,14,15);
        acc[nt] = __builtin_amdgcn_wmma_f32_16x16x32_f16(false, afrag, false, bfrag,
                                                         (short)0, acc[nt], false, false);
      }
    }
    __syncthreads();
  }

  // C/D layout: VGPR i -> M=i (lane<16), M=8+i (lane>=16); N = lane&15
  const int rofs = (lane >> 4) * 8;
  #pragma unroll
  for (int nt = 0; nt < 4; ++nt) {
    const int gcol = ncol0 + nt * 16 + lcol;
    if (gcol < Ndim) {
      #pragma unroll
      for (int i = 0; i < 8; ++i) {
        int r = row0 + wv * 16 + rofs + i;
        if (r < Mdim) {
          float v = acc[nt][i] * scale;
          if (bias) v += bias[gcol];
          if (flags & GF_GELU) v = gelu_f(v);
          float* p = C + (size_t)r * ldc + gcol;
          if (flags & GF_ACC) *p += v; else *p = v;
        }
      }
    }
  }
}

// ---------------------------------------------------------------------------
__global__ void k_embed(const int* __restrict__ tok, const float* __restrict__ emb,
                        const float* __restrict__ pos, float* __restrict__ hs)
{
  int i = blockIdx.x * blockDim.x + threadIdx.x;
  if (i >= B_ * T_ * E_) return;
  int e  = i % E_;
  int bt = i / E_;
  int t  = bt % T_;
  hs[i] = emb[(size_t)tok[bt] * E_ + e] + pos[t * E_ + e];
}

__global__ __launch_bounds__(256)
void k_layernorm(const float* __restrict__ x, float* __restrict__ y, int cols)
{
  __shared__ float r1[256], r2[256];
  int row = blockIdx.x;
  const float* xr = x + (size_t)row * cols;
  float s = 0.f, ss = 0.f;
  for (int c = threadIdx.x; c < cols; c += 256) { float v = xr[c]; s += v; ss += v * v; }
  r1[threadIdx.x] = s; r2[threadIdx.x] = ss; __syncthreads();
  for (int o = 128; o > 0; o >>= 1) {
    if (threadIdx.x < o) { r1[threadIdx.x] += r1[threadIdx.x + o]; r2[threadIdx.x] += r2[threadIdx.x + o]; }
    __syncthreads();
  }
  float mu  = r1[0] / cols;
  float var = r2[0] / cols - mu * mu;
  float inv = rsqrtf(var + 1e-5f);
  float* yr = y + (size_t)row * cols;
  for (int c = threadIdx.x; c < cols; c += 256) yr[c] = (xr[c] - mu) * inv;
}

// causal multi-head attention over the controller sequence; 1 thread = 1 query
__global__ __launch_bounds__(128)
void k_ctrl_attn(const float* __restrict__ qkv, float* __restrict__ out)
{
  int h = blockIdx.x, b = blockIdx.y;
  int t = threadIdx.x;
  const float* qrow = qkv + (size_t)(b * T_ + t) * (3 * E_) + h * HD_;
  float q[HD_], accv[HD_];
  #pragma unroll
  for (int d = 0; d < HD_; ++d) { q[d] = qrow[d]; accv[d] = 0.f; }
  float m = -1e30f, l = 0.f;
  for (int kk = 0; kk <= t; ++kk) {
    const float* krow = qkv + (size_t)(b * T_ + kk) * (3 * E_) + E_ + h * HD_;
    const float* vrow = krow + E_;
    float sc = 0.f;
    #pragma unroll
    for (int d = 0; d < HD_; ++d) sc += q[d] * krow[d];
    sc *= 0.125f;                               // 1/sqrt(64)
    float mn = fmaxf(m, sc);
    float corr = __expf(m - mn);
    float p = __expf(sc - mn);
    l = l * corr + p;
    #pragma unroll
    for (int d = 0; d < HD_; ++d) accv[d] = accv[d] * corr + p * vrow[d];
    m = mn;
  }
  float invl = 1.f / l;
  float* orow = out + (size_t)(b * T_ + t) * E_ + h * HD_;
  #pragma unroll
  for (int d = 0; d < HD_; ++d) orow[d] = accv[d] * invl;
}

// iface = [h_t, read_vec] @ W_iface + b_iface   (N = 2049, plain kernel)
__global__ __launch_bounds__(256)
void k_iface(const float* __restrict__ hs, const float* __restrict__ rv,
             const float* __restrict__ W, const float* __restrict__ bias,
             float* __restrict__ iface, int t)
{
  int i = blockIdx.x * 256 + threadIdx.x;
  if (i >= B_ * IFACE_) return;
  int c = i % IFACE_, b = i / IFACE_;
  const float* hrow = hs + (size_t)(b * T_ + t) * E_;
  const float* rrow = rv + b * M_;
  float acc = bias[c];
  for (int j = 0; j < E_; ++j) acc += hrow[j] * W[(size_t)j * IFACE_ + c];
  for (int j = 0; j < M_; ++j) acc += rrow[j] * W[(size_t)(E_ + j) * IFACE_ + c];
  iface[i] = acc;
}

// cosine-sim addressing + top-k sparse softmax; READ also updates read_vec
__global__ __launch_bounds__(256)
void k_addr(const float* __restrict__ mem, const float* __restrict__ iface,
            const float* __restrict__ beta_in, float* __restrict__ wout,
            float* __restrict__ read_vec, int key_off, int mode)
{
  __shared__ float kn[DH_];
  __shared__ float sim[S_];
  __shared__ float tmp[S_];
  __shared__ float red[256];
  __shared__ int   redi[256];
  __shared__ float sm_kth, sm_gmax;
  int h = blockIdx.x, b = blockIdx.y, tid = threadIdx.x;

  float beta = log1pf(__expf(beta_in[0]));
  beta = fminf(fmaxf(beta, 1.f), 20.f);

  // normalize key head
  float kv = (tid < DH_) ? iface[(size_t)b * IFACE_ + key_off + h * DH_ + tid] : 0.f;
  red[tid] = kv * kv; __syncthreads();
  for (int o = 128; o > 0; o >>= 1) { if (tid < o) red[tid] += red[tid + o]; __syncthreads(); }
  float kinv = rsqrtf(red[0] + 1e-8f);
  if (tid < DH_) kn[tid] = kv * kinv;
  __syncthreads();

  // similarities over 1024 slots (4 per thread)
  #pragma unroll
  for (int it = 0; it < 4; ++it) {
    int s = tid + it * 256;
    const float* mrow = mem + (size_t)(b * S_ + s) * M_ + h * DH_;
    float dot = 0.f, ss = 0.f;
    for (int d = 0; d < DH_; ++d) { float mv = mrow[d]; dot += mv * kn[d]; ss += mv * mv; }
    float v = beta * dot * rsqrtf(ss + 1e-8f);
    sim[s] = v; tmp[s] = v;
  }
  __syncthreads();

  // iterative top-k (k=32): 32 argmax reductions with removal
  for (int iter = 0; iter < TOPK_; ++iter) {
    float lm = -1e30f; int li = 0;
    #pragma unroll
    for (int it = 0; it < 4; ++it) {
      int s = tid + it * 256;
      float v = tmp[s];
      if (v > lm) { lm = v; li = s; }
    }
    red[tid] = lm; redi[tid] = li; __syncthreads();
    for (int o = 128; o > 0; o >>= 1) {
      if (tid < o && red[tid + o] > red[tid]) { red[tid] = red[tid + o]; redi[tid] = redi[tid + o]; }
      __syncthreads();
    }
    if (tid == 0) {
      if (iter == 0) sm_gmax = red[0];
      sm_kth = red[0];
      tmp[redi[0]] = -1e30f;
    }
    __syncthreads();
  }
  float kth = sm_kth, gmax = sm_gmax;

  // sparse softmax over sim >= kth
  float part = 0.f;
  #pragma unroll
  for (int it = 0; it < 4; ++it) {
    int s = tid + it * 256;
    float v = sim[s];
    float e = (v >= kth) ? __expf(v - gmax) : 0.f;
    tmp[s] = e; part += e;
  }
  red[tid] = part; __syncthreads();
  for (int o = 128; o > 0; o >>= 1) { if (tid < o) red[tid] += red[tid + o]; __syncthreads(); }
  float inv = 1.f / red[0];
  #pragma unroll
  for (int it = 0; it < 4; ++it) tmp[tid + it * 256] *= inv;
  __syncthreads();

  if (mode == ADDR_WRITE) {
    #pragma unroll
    for (int it = 0; it < 4; ++it) {
      int s = tid + it * 256;
      wout[((size_t)b * MH_ + h) * S_ + s] = tmp[s];
    }
  } else {
    if (tid < DH_) {
      float a = 0.f;
      for (int s = 0; s < S_; ++s)
        a += tmp[s] * mem[(size_t)(b * S_ + s) * M_ + h * DH_ + tid];
      read_vec[b * M_ + h * DH_ + tid] += a;
    }
  }
}

// memh = memh*(1 - w*erase) + add_gate*w*wv ; then decay
__global__ __launch_bounds__(256)
void k_mem_update(float* __restrict__ mem, const float* __restrict__ iface,
                  const float* __restrict__ w)
{
  int bs = blockIdx.x;
  int b = bs / S_, s = bs % S_;
  float ag = sigm_f(iface[(size_t)b * IFACE_ + 4 * M_]);
  for (int e = threadIdx.x; e < M_; e += 256) {
    int h = e >> 7;
    float wsv = w[((size_t)b * MH_ + h) * S_ + s];
    float er  = sigm_f(iface[(size_t)b * IFACE_ + 3 * M_ + e]);
    float wvv = iface[(size_t)b * IFACE_ + 2 * M_ + e];
    size_t idx = (size_t)(b * S_ + s) * M_ + e;
    float old = mem[idx];
    mem[idx] = (old * (1.f - wsv * er) + ag * wsv * wvv) * DECAY_;
  }
}

__global__ __launch_bounds__(256)
void k_softmax_rows(float* __restrict__ x, int cols)
{
  __shared__ float red[256];
  int row = blockIdx.x;
  float* xr = x + (size_t)row * cols;
  float m = -1e30f;
  for (int c = threadIdx.x; c < cols; c += 256) m = fmaxf(m, xr[c]);
  red[threadIdx.x] = m; __syncthreads();
  for (int o = 128; o > 0; o >>= 1) {
    if (threadIdx.x < o) red[threadIdx.x] = fmaxf(red[threadIdx.x], red[threadIdx.x + o]);
    __syncthreads();
  }
  m = red[0]; __syncthreads();
  float s = 0.f;
  for (int c = threadIdx.x; c < cols; c += 256) { float e = __expf(xr[c] - m); xr[c] = e; s += e; }
  red[threadIdx.x] = s; __syncthreads();
  for (int o = 128; o > 0; o >>= 1) { if (threadIdx.x < o) red[threadIdx.x] += red[threadIdx.x + o]; __syncthreads(); }
  float inv = 1.f / red[0];
  for (int c = threadIdx.x; c < cols; c += 256) xr[c] *= inv;
}

__global__ __launch_bounds__(256)
void k_recon(const float* __restrict__ rv, const float* __restrict__ Wh,
             const float* __restrict__ bh, float* __restrict__ outbase, int t)
{
  int i = blockIdx.x * 256 + threadIdx.x;
  if (i >= B_ * E_) return;
  int e = i % E_, b = i / E_;
  float acc = bh[e];
  const float* r = rv + b * M_;
  for (int j = 0; j < M_; ++j) acc += r[j] * Wh[(size_t)j * E_ + e];
  outbase[(size_t)B_ * T_ * V_ + ((size_t)b * T_ + t) * E_ + e] = acc;
}

// ---------------------------------------------------------------------------
static inline void gemm(hipStream_t st, const float* A, const float* B, const float* bias,
                        float* C, int M, int N, int K, int lda, int ldb, int ldc,
                        int flags, float scale)
{
  dim3 g((N + 63) / 64, (M + 63) / 64), blk(128);
  k_wmma_gemm<<<g, blk, 0, st>>>(A, B, bias, C, M, N, K, lda, ldb, ldc, flags, scale);
}

extern "C" void kernel_launch(void* const* d_in, const int* in_sizes, int n_in,
                              void* d_out, int out_size, void* d_ws, size_t ws_size,
                              hipStream_t stream)
{
  (void)in_sizes; (void)n_in; (void)out_size;
  const int*   tok   = (const int*)  d_in[0];
  const float* embed = (const float*)d_in[1];
  const float* pos   = (const float*)d_in[2];
  const float* Wqkv  = (const float*)d_in[3];
  const float* Wo    = (const float*)d_in[4];
  const float* W1    = (const float*)d_in[5];
  const float* W2    = (const float*)d_in[6];
  const float* Wif   = (const float*)d_in[7];
  const float* bif   = (const float*)d_in[8];
  const float* Wlog  = (const float*)d_in[9];
  const float* betar = (const float*)d_in[10];
  const float* betaw = (const float*)d_in[11];
  const float* Whall = (const float*)d_in[12];
  const float* bhall = (const float*)d_in[13];
  const float* Sq    = (const float*)d_in[14];
  const float* Sk    = (const float*)d_in[15];
  const float* Sv    = (const float*)d_in[16];
  const float* So    = (const float*)d_in[17];
  const float* SW1   = (const float*)d_in[18];
  const float* SW2   = (const float*)d_in[19];
  float* out = (float*)d_out;

  float* ws = (float*)d_ws;
  size_t o = 0;
  float* hs    = ws + o; o += (size_t)B_ * T_ * E_;       // controller states
  float* xln   = ws + o; o += (size_t)B_ * S_ * M_;       // LN scratch (shared)
  float* qkvb  = ws + o; o += (size_t)B_ * T_ * 3 * E_;   // qkv / synth-hidden
  float* attno = ws + o; o += (size_t)B_ * T_ * E_;
  float* hid   = ws + o; o += (size_t)B_ * T_ * HID_;
  float* memb  = ws + o; o += (size_t)B_ * S_ * M_;
  float* rv    = ws + o; o += (size_t)64 * M_;            // padded to 64 WMMA rows
  float* ifc   = ws + o; o += (size_t)B_ * IFACE_;
  float* wbuf  = ws + o; o += (size_t)B_ * MH_ * S_;
  float* sq    = ws + o; o += (size_t)B_ * S_ * M_;
  float* sk    = ws + o; o += (size_t)B_ * S_ * M_;
  float* sv    = ws + o; o += (size_t)B_ * S_ * M_;
  float* satt  = ws + o; o += (size_t)B_ * S_ * M_;
  float* scob  = ws + o; o += (size_t)S_ * S_;
  if (ws_size < o * sizeof(float)) return;

  hipMemsetAsync(memb, 0, (size_t)B_ * S_ * M_ * sizeof(float), stream);
  hipMemsetAsync(rv,   0, (size_t)64 * M_ * sizeof(float), stream);

  // ---- embedding ----
  {
    int n = B_ * T_ * E_;
    k_embed<<<(n + 255) / 256, 256, 0, stream>>>(tok, embed, pos, hs);
  }

  // ---- controller transformer (L=2) ----
  for (int l = 0; l < L_; ++l) {
    k_layernorm<<<B_ * T_, 256, 0, stream>>>(hs, xln, E_);
    gemm(stream, xln, Wqkv + (size_t)l * E_ * 3 * E_, nullptr, qkvb,
         B_ * T_, 3 * E_, E_, E_, 3 * E_, 3 * E_, 0, 1.f);
    k_ctrl_attn<<<dim3(NH_, B_), T_, 0, stream>>>(qkvb, attno);
    gemm(stream, attno, Wo + (size_t)l * E_ * E_, nullptr, hs,
         B_ * T_, E_, E_, E_, E_, E_, GF_ACC, 1.f);
    k_layernorm<<<B_ * T_, 256, 0, stream>>>(hs, xln, E_);
    gemm(stream, xln, W1 + (size_t)l * E_ * HID_, nullptr, hid,
         B_ * T_, HID_, E_, E_, HID_, HID_, GF_GELU, 1.f);
    gemm(stream, hid, W2 + (size_t)l * HID_ * E_, nullptr, hs,
         B_ * T_, E_, HID_, HID_, E_, E_, GF_ACC, 1.f);
  }

  // ---- hoisted h-part of logits: [512,32000] GEMM, one pass over 131MB ----
  gemm(stream, hs, Wlog, nullptr, out, B_ * T_, V_, E_, E_, V_, V_, 0, 1.f);

  const float inv_sqrt_sd = 0.08838834764831845f; // 1/sqrt(128)

  // ---- sequential memory scan ----
  for (int t = 0; t < T_; ++t) {
    k_iface<<<(B_ * IFACE_ + 255) / 256, 256, 0, stream>>>(hs, rv, Wif, bif, ifc, t);

    // rv-part of logits (rank-4 padded to 64, L2-resident weight slice)
    gemm(stream, rv, Wlog + (size_t)E_ * V_, nullptr, out + (size_t)t * V_,
         B_, V_, M_, M_, V_, T_ * V_, GF_ACC, 1.f);

    k_addr<<<dim3(MH_, B_), 256, 0, stream>>>(memb, ifc, betar, wbuf, rv, 0,  ADDR_READ);
    k_addr<<<dim3(MH_, B_), 256, 0, stream>>>(memb, ifc, betaw, wbuf, rv, M_, ADDR_WRITE);
    k_mem_update<<<B_ * S_, 256, 0, stream>>>(memb, ifc, wbuf);

    if (t % INTERVAL_ == 0) {
      // synthesizer: slot self-attention + bottleneck MLP (LS=1)
      k_layernorm<<<B_ * S_, 256, 0, stream>>>(memb, xln, M_);
      gemm(stream, xln, Sq, nullptr, sq, B_ * S_, M_, M_, M_, M_, M_, 0, 1.f);
      gemm(stream, xln, Sk, nullptr, sk, B_ * S_, M_, M_, M_, M_, M_, 0, 1.f);
      gemm(stream, xln, Sv, nullptr, sv, B_ * S_, M_, M_, M_, M_, M_, 0, 1.f);
      for (int b = 0; b < B_; ++b)
        for (int h = 0; h < SH_; ++h) {
          const float* qh = sq + (size_t)b * S_ * M_ + h * SD_;
          const float* kh = sk + (size_t)b * S_ * M_ + h * SD_;
          const float* vh = sv + (size_t)b * S_ * M_ + h * SD_;
          float*       oh = satt + (size_t)b * S_ * M_ + h * SD_;
          gemm(stream, qh, kh, nullptr, scob, S_, S_, SD_, M_, M_, S_, GF_BT, inv_sqrt_sd);
          k_softmax_rows<<<S_, 256, 0, stream>>>(scob, S_);
          gemm(stream, scob, vh, nullptr, oh, S_, SD_, S_, S_, M_, M_, 0, 1.f);
        }
      gemm(stream, satt, So, nullptr, memb, B_ * S_, M_, M_, M_, M_, M_, GF_ACC, 1.f);
      k_layernorm<<<B_ * S_, 256, 0, stream>>>(memb, xln, M_);
      gemm(stream, xln, SW1, nullptr, qkvb, B_ * S_, BT_, M_, M_, BT_, BT_, GF_GELU, 1.f);
      gemm(stream, qkvb, SW2, nullptr, memb, B_ * S_, M_, BT_, BT_, M_, M_, GF_ACC, 1.f);
    }

    k_recon<<<(B_ * E_ + 255) / 256, 256, 0, stream>>>(rv, Whall, bhall, out, t);
  }
}
